// MultiAxisKernelCore_15848429322482
// MI455X (gfx1250) — matrix operand order repaired and checked by reading, not compiled, not run on hardware
//
#include <hip/hip_runtime.h>
#include <hip/hip_bf16.h>
#include <math.h>

#define PDIM 64
#define NBASES 8
#define TAPS 169     // 13x13
#define HIDDEN 32
#define BATCH 8
#define CH 96
#define HW 65536     // 256*256
#define TILE 32
#define HALO 6
#define LTILE 44     // 32 + 2*6

typedef __attribute__((ext_vector_type(16))) __bf16 v16bf;
typedef __attribute__((ext_vector_type(8)))  float  v8f;
typedef __attribute__((ext_vector_type(4)))  float  f4;   // raw vector (NT-builtin friendly)

#if defined(__AMDGCN__) && __has_builtin(__builtin_amdgcn_global_load_async_to_lds_b32)
#define ASYNC_LDS 1
typedef __attribute__((address_space(1))) int GI;   // builtin expects int-typed AS pointers
typedef __attribute__((address_space(3))) int LI;
#endif

// ---------------------------------------------------------------- pooled mean
__global__ void dk_pool(const float* __restrict__ x, float* __restrict__ pooled) {
  const int bc = blockIdx.x;            // 0..511 : b*64 + c
  const int b = bc >> 6, c = bc & 63;
  const float* plane = x + ((size_t)(b * CH + c)) * HW;
  float s = 0.f;
  for (int i = threadIdx.x; i < HW; i += 256)
    s += __builtin_nontemporal_load(&plane[i]);        // streaming: keep L2 for fused_t
  __shared__ float red[256];
  red[threadIdx.x] = s;
  __syncthreads();
  for (int off = 128; off > 0; off >>= 1) {
    if (threadIdx.x < off) red[threadIdx.x] += red[threadIdx.x + off];
    __syncthreads();
  }
  if (threadIdx.x == 0) pooled[bc] = red[0] * (1.0f / 65536.0f);
}

// ------------------------------------------- MLPs + combined per-(b,c) kernel
__global__ void dk_mlp_comb(const float* __restrict__ pooled,
                            const float* __restrict__ lk_bases,
                            const float* __restrict__ rw1, const float* __restrict__ rb1,
                            const float* __restrict__ rw2, const float* __restrict__ rb2,
                            const float* __restrict__ sw1, const float* __restrict__ sb1,
                            const float* __restrict__ sw2, const float* __restrict__ sb2,
                            const float* __restrict__ gw1, const float* __restrict__ gb1,
                            const float* __restrict__ gw2, const float* __restrict__ gb2,
                            float* __restrict__ comb) {
  const int b = blockIdx.x;             // 0..7
  const int tid = threadIdx.x;
  __shared__ float sp[PDIM];
  __shared__ float sh[3 * HIDDEN];      // router / sk / gate hiddens
  __shared__ float srw[PDIM * NBASES];
  __shared__ float sdyn[PDIM * 9];
  __shared__ float sglk[PDIM], sgdyn[PDIM];

  if (tid < PDIM) sp[tid] = pooled[b * PDIM + tid];
  __syncthreads();

  if (tid < 3 * HIDDEN) {               // hidden layers, exact gelu
    const int which = tid >> 5, j = tid & 31;
    const float* w1 = (which == 0) ? rw1 : (which == 1) ? sw1 : gw1;
    const float* b1 = (which == 0) ? rb1 : (which == 1) ? sb1 : gb1;
    float a = b1[j];
    for (int i = 0; i < PDIM; ++i) a = fmaf(w1[j * PDIM + i], sp[i], a);
    sh[tid] = 0.5f * a * (1.0f + erff(a * 0.70710678118f));
  }
  __syncthreads();

  if (tid < PDIM) {
    const int c = tid;
    const float* hr = sh;
    const float* hs = sh + HIDDEN;
    const float* hg = sh + 2 * HIDDEN;
    // router logits -> softmax over 8 bases
    float lg[NBASES]; float mx = -1e30f;
    for (int n = 0; n < NBASES; ++n) {
      float a = rb2[c * NBASES + n];
      for (int j = 0; j < HIDDEN; ++j) a = fmaf(rw2[(c * NBASES + n) * HIDDEN + j], hr[j], a);
      lg[n] = a; mx = fmaxf(mx, a);
    }
    float den = 0.f;
    for (int n = 0; n < NBASES; ++n) { lg[n] = __expf(lg[n] - mx); den += lg[n]; }
    const float inv = 1.0f / den;
    for (int n = 0; n < NBASES; ++n) srw[c * NBASES + n] = lg[n] * inv;
    // dynamic 3x3 kernel
    for (int t = 0; t < 9; ++t) {
      float a = sb2[c * 9 + t];
      for (int j = 0; j < HIDDEN; ++j) a = fmaf(sw2[(c * 9 + t) * HIDDEN + j], hs[j], a);
      sdyn[c * 9 + t] = a;
    }
    // gate softmax over {lk, dyn}
    float a0 = gb2[c], a1 = gb2[PDIM + c];
    for (int j = 0; j < HIDDEN; ++j) {
      a0 = fmaf(gw2[c * HIDDEN + j], hg[j], a0);
      a1 = fmaf(gw2[(PDIM + c) * HIDDEN + j], hg[j], a1);
    }
    const float m = fmaxf(a0, a1);
    const float e0 = __expf(a0 - m), e1 = __expf(a1 - m), d = e0 + e1;
    sglk[c] = e0 / d; sgdyn[c] = e1 / d;
  }
  __syncthreads();

  // comb = g_lk * (rw . bases)  +  g_dyn * dyn embedded at taps 5..7
  for (int idx = tid; idx < PDIM * TAPS; idx += 256) {
    const int c = idx / TAPS, tap = idx - c * TAPS;
    float s = 0.f;
    for (int n = 0; n < NBASES; ++n) s = fmaf(srw[c * NBASES + n], lk_bases[n * TAPS + tap], s);
    float v = sglk[c] * s;
    const int ky = tap / 13, kx = tap - ky * 13;
    if (ky >= 5 && ky <= 7 && kx >= 5 && kx <= 7)
      v = fmaf(sgdyn[c], sdyn[c * 9 + (ky - 5) * 3 + (kx - 5)], v);
    comb[b * PDIM * TAPS + idx] = v;
  }
}

// ----------------------- fused 13x13 depthwise conv, write pixel-major bf16
__global__ void __launch_bounds__(256) dk_conv(const float* __restrict__ x,
                                               const float* __restrict__ comb,
                                               __bf16* __restrict__ fused_t) {
  const int blk = blockIdx.x;
  const int tile = blk & 63;            // 8x8 tiles of 32x32
  const int bc = blk >> 6;              // 0..511
  const int b = bc >> 6, c = bc & 63;
  const int ty = tile >> 3, tx = tile & 7;
  __shared__ float stile[LTILE * LTILE];
  __shared__ float skern[TAPS];
  const float* plane = x + ((size_t)(b * CH + c)) * HW;
  const int tid = threadIdx.x;

#if defined(ASYNC_LDS)
  if (ty >= 1 && ty <= 6 && tx >= 1 && tx <= 6) {
    // Fully-interior tile: ASYNCcnt-tracked direct global->LDS DMA, no VGPR staging.
    const float* src0 = plane + (ty * TILE - HALO) * 256 + (tx * TILE - HALO);
    for (int i = tid; i < LTILE * LTILE; i += 256) {
      const int ly = i / LTILE, lx = i - ly * LTILE;
      __builtin_amdgcn_global_load_async_to_lds_b32(
          (GI*)(uintptr_t)(src0 + ly * 256 + lx),
          (LI*)(uintptr_t)(stile + i), 0, 0);
    }
    asm volatile("s_wait_asynccnt 0x0" ::: "memory");
  } else
#endif
  {
    for (int i = tid; i < LTILE * LTILE; i += 256) {
      const int ly = i / LTILE, lx = i - ly * LTILE;
      const int gy = ty * TILE - HALO + ly, gx = tx * TILE - HALO + lx;
      float v = 0.f;
      if ((unsigned)gy < 256u && (unsigned)gx < 256u)
        v = __builtin_nontemporal_load(&plane[gy * 256 + gx]);
      stile[i] = v;
    }
  }
  if (tid < TAPS) skern[tid] = comb[bc * TAPS + tid];
  __syncthreads();

  for (int r = 0; r < 4; ++r) {
    const int oidx = r * 256 + tid;     // 32x32 = 1024 outputs / block
    const int oy = oidx >> 5, ox = oidx & 31;
    float acc = 0.f;
#pragma unroll
    for (int ky = 0; ky < 13; ++ky) {
      const float* row = &stile[(oy + ky) * LTILE + ox];
#pragma unroll
      for (int kx = 0; kx < 13; ++kx) acc = fmaf(row[kx], skern[ky * 13 + kx], acc);
    }
    const int p = (ty * TILE + oy) * 256 + (tx * TILE + ox);
    // fused_t stays RT: 64 MB fits in the 192 MB L2 and is re-read by dk_mix_wmma.
    fused_t[((size_t)(b * HW + p)) * PDIM + c] = (__bf16)acc;
  }
}

// ------------------- 64x64 channel mix via v_wmma_f32_16x16x32_bf16 + residual
__global__ void __launch_bounds__(256) dk_mix_wmma(const float* __restrict__ x,
                                                   const __bf16* __restrict__ fused_t,
                                                   const float* __restrict__ fuse_w,
                                                   const float* __restrict__ fuse_b,
                                                   float* __restrict__ out) {
  const int lane = threadIdx.x & 31;
  const int wid = threadIdx.x >> 5;
  const int gwave = blockIdx.x * 8 + wid;   // 0..2047
  const int lh = lane >> 4;                 // lane half (0/1)
  const int ln = lane & 15;

  // Preload A (fuse_w as bf16) in contiguous K order (k = 16*lh + j): 4 M-tiles x 2 K-halves.
  v16bf Aw[4][2];
#pragma unroll
  for (int mt = 0; mt < 4; ++mt)
#pragma unroll
    for (int kh = 0; kh < 2; ++kh) {
      v16bf a;
#pragma unroll
      for (int j = 0; j < 16; ++j) {
        const int kl = lh * 16 + j;
        a[j] = (__bf16)fuse_w[(mt * 16 + ln) * PDIM + kh * 32 + kl];
      }
      Aw[mt][kh] = a;
    }

#pragma unroll
  for (int t = 0; t < 16; ++t) {            // 16 pixel-tiles (of 16 px) per wave
    const int tile = gwave * 16 + t;        // 0..32767
    const int pix0 = tile * 16;
    const int b = pix0 >> 16;
    const int p = (pix0 & 65535) + ln;
    // B operand: 16 contiguous bf16 channels for this pixel (pixel-major layout).
    // Regular-temporal loads: these should hit L2 (fused_t resident).
    const __bf16* bp = fused_t + ((size_t)(b * HW + p)) * PDIM;
    const v16bf Bv0 = *(const v16bf*)(bp + lh * 16);        // K = 0..31
    const v16bf Bv1 = *(const v16bf*)(bp + 32 + lh * 16);   // K = 32..63
#pragma unroll
    for (int mt = 0; mt < 4; ++mt) {
      v8f acc = {};
      acc = __builtin_amdgcn_wmma_f32_16x16x32_bf16(false, Aw[mt][0], false, Bv0,
                                                    (short)0, acc, false, false);
      acc = __builtin_amdgcn_wmma_f32_16x16x32_bf16(false, Aw[mt][1], false, Bv1,
                                                    (short)0, acc, false, false);
#pragma unroll
      for (int r = 0; r < 8; ++r) {
        const int o = mt * 16 + r + (lh ? 8 : 0);
        const size_t oi = ((size_t)(b * CH + o)) * HW + p;
        const float v = acc[r] + fuse_b[o] + __builtin_nontemporal_load(&x[oi]);
        __builtin_nontemporal_store(v, &out[oi]);   // never re-read: stream past L2
      }
    }
  }
}

// ------------------------------------------------------------- x2 passthrough
__global__ void dk_copy_x2(const f4* __restrict__ x4, f4* __restrict__ out4) {
  const size_t i = (size_t)blockIdx.x * 256 + threadIdx.x;   // 4,194,304 total
  const size_t per_b = (size_t)32 * HW / 4;                  // 524288
  const size_t b = i / per_b, rem = i - b * per_b;
  const size_t idx = ((b * CH + PDIM) * (size_t)HW) / 4 + rem;
  const f4 v = __builtin_nontemporal_load(&x4[idx]);
  __builtin_nontemporal_store(v, &out4[idx]);
}

extern "C" void kernel_launch(void* const* d_in, const int* in_sizes, int n_in,
                              void* d_out, int out_size, void* d_ws, size_t ws_size,
                              hipStream_t stream) {
  const float* x   = (const float*)d_in[0];
  const float* lk  = (const float*)d_in[1];
  const float* rw1 = (const float*)d_in[2];
  const float* rb1 = (const float*)d_in[3];
  const float* rw2 = (const float*)d_in[4];
  const float* rb2 = (const float*)d_in[5];
  const float* sw1 = (const float*)d_in[6];
  const float* sb1 = (const float*)d_in[7];
  const float* sw2 = (const float*)d_in[8];
  const float* sb2 = (const float*)d_in[9];
  const float* gw1 = (const float*)d_in[10];
  const float* gb1 = (const float*)d_in[11];
  const float* gw2 = (const float*)d_in[12];
  const float* gb2 = (const float*)d_in[13];
  const float* fw  = (const float*)d_in[14];
  const float* fb  = (const float*)d_in[15];
  float* out = (float*)d_out;

  float* wsf = (float*)d_ws;
  float* pooled = wsf;                                  // 512 f32
  float* comb = wsf + 512;                              // 8*64*169 f32
  __bf16* fused_t = (__bf16*)((char*)d_ws + (1u << 20)); // 64 MB pixel-major bf16

  dk_pool<<<BATCH * PDIM, 256, 0, stream>>>(x, pooled);
  dk_mlp_comb<<<BATCH, 256, 0, stream>>>(pooled, lk, rw1, rb1, rw2, rb2,
                                         sw1, sb1, sw2, sb2, gw1, gb1, gw2, gb2, comb);
  dk_conv<<<BATCH * PDIM * 64, 256, 0, stream>>>(x, comb, fused_t);
  dk_mix_wmma<<<256, 256, 0, stream>>>(x, fused_t, fw, fb, out);
  dk_copy_x2<<<16384, 256, 0, stream>>>((const f4*)x, (f4*)out);
}
